// KuraNet_52338471469173
// MI455X (gfx1250) — matrix operand order repaired and checked by reading, not compiled, not run on hardware
//
#include <hip/hip_runtime.h>
#include <hip/hip_bf16.h>

#define NN 1024
#define FDIM 16
#define HD 128
#define NP (NN*NN)
#define NHW (NN*HD)
#define STEPS 150
#define ALPHA 0.1f
#define SLOPE 0.01f
#define BN_EPS 1e-5f

typedef __attribute__((ext_vector_type(16))) _Float16 v16h;
typedef __attribute__((ext_vector_type(8)))  float    v8f;
typedef __attribute__((ext_vector_type(4)))  int      v4i;

#if __has_builtin(__builtin_amdgcn_global_load_async_to_lds_b128) && \
    __has_builtin(__builtin_amdgcn_s_wait_asynccnt)
#define HAVE_ASYNC_LDS 1
typedef __attribute__((address_space(1))) v4i* v4i_gp;   // global int4*
typedef __attribute__((address_space(3))) v4i* v4i_lp;   // LDS int4*
#else
#define HAVE_ASYNC_LDS 0
#endif

// ---- workspace layout (float offsets) ----
#define OFF_KLOG  0                       // [NP]  raw logits
#define OFF_KFIN  (NP)                    // [NP]  final symmetric K
#define OFF_UB    (2*NP)                  // [NHW] a1*u + t1
#define OFF_VB    (2*NP + NHW)            // [NHW] a1*v
#define OFF_STATS (2*NP + 2*NHW)          // [4H]  sum_u, sq_u, sum_v, sq_v
#define OFF_S2    (OFF_STATS + 4*HD)      // [2H]  sum2, sq2 (BN2)
#define OFF_AFF   (OFF_S2 + 2*HD)         // [2H]  a2, c2
#define OFF_T1    (OFF_AFF + 2*HD)        // [2H]  a1, t1
#define OFF_PART  (OFF_T1 + 2*HD)         // [1024] partial reductions
#define OFF_MZ    (OFF_PART + 1024)       // [2]   max, Z
#define OFF_TH    (OFF_MZ + 2)            // [2N]  theta ping-pong
#define OFF_BAR   (OFF_TH + 2*NN)         // [4]   grid barrier (uints)
#define ZLEN      (OFF_BAR + 4 - OFF_STATS)

__device__ __forceinline__ float lrelu(float v){ return v > 0.f ? v : SLOPE*v; }

// ---------- init: zero all accumulators / theta / barrier each call ----------
__global__ void k_zero(float* ws){
  for (int i = threadIdx.x; i < ZLEN; i += blockDim.x) ws[OFF_STATS + i] = 0.f;
}

// ---------- u = x@w1[:16], v = x@w1[16:], + BN1 stats ----------
__global__ void k_uv(const float* __restrict__ x, const float* __restrict__ w1, float* ws){
  __shared__ float xs[FDIM];
  int i = blockIdx.x, c = threadIdx.x;
  if (c < FDIM) xs[c] = x[i*FDIM + c];
  __syncthreads();
  float u = 0.f, v = 0.f;
#pragma unroll
  for (int f = 0; f < FDIM; f++){
    float xv = xs[f];
    u += xv * w1[f*HD + c];
    v += xv * w1[(FDIM + f)*HD + c];
  }
  ws[OFF_UB + i*HD + c] = u;
  ws[OFF_VB + i*HD + c] = v;
  atomicAdd(&ws[OFF_STATS + c],        u);
  atomicAdd(&ws[OFF_STATS + HD + c],   u*u);
  atomicAdd(&ws[OFF_STATS + 2*HD + c], v);
  atomicAdd(&ws[OFF_STATS + 3*HD + c], v*v);
}

// ---------- BN1 affine coefficients ----------
__global__ void k_bn1a(const float* b1, const float* g1, const float* be1, float* ws){
  int c = threadIdx.x;
  float su = ws[OFF_STATS + c], squ = ws[OFF_STATS + HD + c];
  float sv = ws[OFF_STATS + 2*HD + c], sqv = ws[OFF_STATS + 3*HD + c];
  float mu_u = su*(1.f/NN), mu_v = sv*(1.f/NN);
  float var = (squ*(1.f/NN) - mu_u*mu_u) + (sqv*(1.f/NN) - mu_v*mu_v);
  float a1 = g1[c] * rsqrtf(var + BN_EPS);
  float mu = mu_u + mu_v + b1[c];
  ws[OFF_T1 + c]      = a1;
  ws[OFF_T1 + HD + c] = a1*(b1[c] - mu) + be1[c];
}

// ---------- fold BN1 into ub / vb ----------
__global__ void k_bn1b(float* ws){
  int i = blockIdx.x, c = threadIdx.x;
  float a1 = ws[OFF_T1 + c], t1 = ws[OFF_T1 + HD + c];
  int idx = i*HD + c;
  ws[OFF_UB + idx] = a1*ws[OFF_UB + idx] + t1;
  ws[OFF_VB + idx] = a1*ws[OFF_VB + idx];
}

// fragment-layout index for w2[k][n] (B of v_wmma_f32_16x16x32_f16):
// frag(kc,nt): lane = ((K_local/8)%2)*16 + n%16 ; half s = K_local%8 + 8*(K_local/16)
__device__ __forceinline__ int bfrag_idx(int k, int n){
  int kc = k >> 5, kr = k & 31, khi = (kr >> 3) & 1, s = (kr & 7) + ((kr >> 4) << 3);
  int nt = n >> 4, ln = n & 15, lane2 = khi*16 + ln;
  return (((kc<<3) + nt)*32 + lane2)*16 + s;
}

// ---------- main fused GEMM: h1 -> h2pre via v_wmma_f32_16x16x32_f16 ----------
// PASS 0: accumulate per-channel sum/sumsq of h2pre (BN2 stats)
// PASS 1: apply BN2 affine + lrelu, dot with w3, write K logits
template<int PASS>
__global__ __launch_bounds__(256) void k_gemm(const float* __restrict__ w2,
                                              const float* __restrict__ w3,
                                              float* ws){
  __shared__ v16h  sB[1024];      // 32KB: w2 pre-swizzled into B-fragment order
  __shared__ float sred[2*HD];
  __shared__ float sAff[3*HD];
#if HAVE_ASYNC_LDS
  __shared__ __align__(16) float sStage[2048];   // 8KB async staging chunk
#endif
  _Float16* sBh = (_Float16*)sB;
  int tid = threadIdx.x;

#if HAVE_ASYNC_LDS
  // stream w2 (f32) into LDS via ASYNCcnt-tracked global_load_async_to_lds_b128,
  // then convert/swizzle LDS->LDS into the f16 B-fragment buffer, 8KB per chunk.
  for (int ch = 0; ch < 8; ch++){
    const float* src = w2 + ch*2048;
    __builtin_amdgcn_global_load_async_to_lds_b128(
        (v4i_gp)(src + tid*4),
        (v4i_lp)(&sStage[tid*4]), 0, 0);
    __builtin_amdgcn_global_load_async_to_lds_b128(
        (v4i_gp)(src + 1024 + tid*4),
        (v4i_lp)(&sStage[1024 + tid*4]), 0, 0);
    __builtin_amdgcn_s_wait_asynccnt(0);
    __syncthreads();
#pragma unroll
    for (int e = 0; e < 8; e++){
      int q = tid*8 + e;                       // 0..2047 within chunk
      int k = ch*16 + (q >> 7), n = q & 127;
      sBh[bfrag_idx(k, n)] = (_Float16)sStage[q];
    }
    __syncthreads();
  }
#else
  for (int e = 0; e < 64; e++){
    int q = tid*64 + e;
    int k = q >> 7, n = q & 127;
    sBh[bfrag_idx(k, n)] = (_Float16)w2[k*HD + n];
  }
#endif
  if (PASS == 1){
    for (int c = tid; c < HD; c += 256){
      sAff[c]        = ws[OFF_AFF + c];
      sAff[HD + c]   = ws[OFF_AFF + HD + c];
      sAff[2*HD + c] = w3[c];
    }
  } else {
    for (int c = tid; c < 2*HD; c += 256) sred[c] = 0.f;
  }
  __syncthreads();

  int wv = tid >> 5, lane = tid & 31;
  int mrow = lane & 15, hi = lane >> 4;
  const float* ub = ws + OFF_UB;
  const float* vb = ws + OFF_VB;

  float csum[8], csq[8];
  if (PASS == 0){
#pragma unroll
    for (int q = 0; q < 8; q++){ csum[q] = 0.f; csq[q] = 0.f; }
  }

  for (int it = 0; it < 8; it++){
    int tt = blockIdx.x*64 + wv*8 + it;          // 65536 tiles total
    int i  = tt >> 6;
    int j0 = (tt & 63) << 4;
    const float* ur = ub + i*HD;
    const float* vr = vb + (j0 + mrow)*HD;

    // A fragments: 16-bit A 16x32 layout; per-lane K = s + 8*hi + 8*(s>>3)
    v16h af[4];
#pragma unroll
    for (int kc = 0; kc < 4; kc++){
      int cb = kc << 5;
#pragma unroll
      for (int s = 0; s < 16; s++){
        int kl = s + hi*8 + ((s >> 3) << 3);
        af[kc][s] = (_Float16)lrelu(ur[cb + kl] + vr[cb + kl]);
      }
    }

    float rowsum[8];
    if (PASS == 1){
#pragma unroll
      for (int r = 0; r < 8; r++) rowsum[r] = 0.f;
    }

#pragma unroll
    for (int nt = 0; nt < 8; nt++){
      v8f acc = {};
#pragma unroll
      for (int kc = 0; kc < 4; kc++){
        v16h bf = sB[((kc<<3) + nt)*32 + lane];
        acc = __builtin_amdgcn_wmma_f32_16x16x32_f16(false, af[kc], false, bf,
                                                     (short)0, acc, false, false);
      }
      if (PASS == 0){
        float s1 = 0.f, s2 = 0.f;
#pragma unroll
        for (int r = 0; r < 8; r++){ float v = acc[r]; s1 += v; s2 += v*v; }
        csum[nt] += s1; csq[nt] += s2;
      } else {
        int n = (nt << 4) + mrow;                 // C layout: column = lane&15
        float a2 = sAff[n], c2 = sAff[HD + n], w3v = sAff[2*HD + n];
#pragma unroll
        for (int r = 0; r < 8; r++){
          rowsum[r] += lrelu(a2*acc[r] + c2) * w3v;
        }
      }
    }

    if (PASS == 1){
#pragma unroll
      for (int r = 0; r < 8; r++){
        float v = rowsum[r];
        v += __shfl_xor(v, 1, 32);
        v += __shfl_xor(v, 2, 32);
        v += __shfl_xor(v, 4, 32);
        v += __shfl_xor(v, 8, 32);
        if (mrow == 0){                           // lane0: rows 0..7, lane16: rows 8..15
          ws[OFF_KLOG + i*NN + j0 + r + hi*8] = v;
        }
      }
    }
  }

  if (PASS == 0){
#pragma unroll
    for (int nt = 0; nt < 8; nt++){
      float s1 = csum[nt] + __shfl_xor(csum[nt], 16, 32);
      float s2 = csq[nt]  + __shfl_xor(csq[nt],  16, 32);
      if (hi == 0){
        atomicAdd(&sred[(nt<<4) + mrow],      s1);
        atomicAdd(&sred[HD + (nt<<4) + mrow], s2);
      }
    }
    __syncthreads();
    for (int c = tid; c < HD; c += 256){
      atomicAdd(&ws[OFF_S2 + c],      sred[c]);
      atomicAdd(&ws[OFF_S2 + HD + c], sred[HD + c]);
    }
  }
}

// ---------- BN2 affine (b2 cancels exactly) ----------
__global__ void k_bn2(const float* g2, const float* be2, float* ws){
  int c = threadIdx.x;
  float s = ws[OFF_S2 + c], sq = ws[OFF_S2 + HD + c];
  float mean = s * (1.f/(float)NP);
  float var  = sq * (1.f/(float)NP) - mean*mean;
  float a2 = g2[c] * rsqrtf(var + BN_EPS);
  ws[OFF_AFF + c]      = a2;
  ws[OFF_AFF + HD + c] = be2[c] - a2*mean;
}

// ---------- softmax reductions over the 1M logits ----------
__global__ void k_maxpart(float* ws){
  __shared__ float sm[256];
  int base = blockIdx.x * 1024;
  float m = -1e30f;
#pragma unroll
  for (int q = 0; q < 4; q++) m = fmaxf(m, ws[OFF_KLOG + base + threadIdx.x + q*256]);
  sm[threadIdx.x] = m; __syncthreads();
  for (int o = 128; o > 0; o >>= 1){
    if (threadIdx.x < o) sm[threadIdx.x] = fmaxf(sm[threadIdx.x], sm[threadIdx.x + o]);
    __syncthreads();
  }
  if (threadIdx.x == 0) ws[OFF_PART + blockIdx.x] = sm[0];
}

__global__ void k_maxfin(float* ws){
  __shared__ float sm[1024];
  sm[threadIdx.x] = ws[OFF_PART + threadIdx.x]; __syncthreads();
  for (int o = 512; o > 0; o >>= 1){
    if (threadIdx.x < o) sm[threadIdx.x] = fmaxf(sm[threadIdx.x], sm[threadIdx.x + o]);
    __syncthreads();
  }
  if (threadIdx.x == 0) ws[OFF_MZ] = sm[0];
}

__global__ void k_exppart(float* ws){
  __shared__ float sm[256];
  float m = ws[OFF_MZ];
  int base = blockIdx.x * 1024;
  float a = 0.f;
#pragma unroll
  for (int q = 0; q < 4; q++) a += __expf(ws[OFF_KLOG + base + threadIdx.x + q*256] - m);
  sm[threadIdx.x] = a; __syncthreads();
  for (int o = 128; o > 0; o >>= 1){
    if (threadIdx.x < o) sm[threadIdx.x] += sm[threadIdx.x + o];
    __syncthreads();
  }
  if (threadIdx.x == 0) ws[OFF_PART + blockIdx.x] = sm[0];
}

__global__ void k_expfin(float* ws){
  __shared__ float sm[1024];
  sm[threadIdx.x] = ws[OFF_PART + threadIdx.x]; __syncthreads();
  for (int o = 512; o > 0; o >>= 1){
    if (threadIdx.x < o) sm[threadIdx.x] += sm[threadIdx.x + o];
    __syncthreads();
  }
  if (threadIdx.x == 0) ws[OFF_MZ + 1] = sm[0];
}

// ---------- K_final = N * 0.5 * (softmax + softmax^T) ----------
__global__ void k_symm(float* ws){
  float m = ws[OFF_MZ], Z = ws[OFF_MZ + 1];
  float sc = ((float)NN * 0.5f) / Z;
  int idx = blockIdx.x*256 + threadIdx.x;
  int i = idx >> 10, j = idx & 1023;
  float e1 = __expf(ws[OFF_KLOG + idx]        - m);
  float e2 = __expf(ws[OFF_KLOG + j*NN + i]   - m);
  ws[OFF_KFIN + idx] = (e1 + e2) * sc;
}

// ---------- persistent Kuramoto integrator with grid barrier ----------
__device__ __forceinline__ float ld_agent(const float* p){
  return __hip_atomic_load(p, __ATOMIC_RELAXED, __HIP_MEMORY_SCOPE_AGENT);
}
__device__ __forceinline__ void st_agent(float* p, float v){
  __hip_atomic_store(p, v, __ATOMIC_RELAXED, __HIP_MEMORY_SCOPE_AGENT);
}

__global__ __launch_bounds__(256) void k_kur(float* ws, float* __restrict__ out){
  __shared__ float ss[NN], cc[NN];
  unsigned* bar = (unsigned*)(ws + OFF_BAR);
  const float* K = ws + OFF_KFIN;
  int tid = threadIdx.x, wv = tid >> 5, lane = tid & 31;

  for (int step = 0; step < STEPS; step++){
    float* thc = ws + OFF_TH + (step & 1)*NN;
    float* thn = ws + OFF_TH + ((step + 1) & 1)*NN;
#pragma unroll
    for (int q = 0; q < 4; q++){
      int j = tid + q*256;
      float t = ld_agent(&thc[j]);
      ss[j] = __sinf(t); cc[j] = __cosf(t);
    }
    __syncthreads();
#pragma unroll
    for (int rr = 0; rr < 4; rr++){
      int i = blockIdx.x*32 + wv*4 + rr;
      const float* Kr = K + i*NN;
      float aS = 0.f, aC = 0.f;
      for (int j = lane; j < NN; j += 32){
        float kv = Kr[j];
        aS += kv * ss[j];
        aC += kv * cc[j];
      }
#pragma unroll
      for (int o = 1; o < 32; o <<= 1){
        aS += __shfl_xor(aS, o, 32);
        aC += __shfl_xor(aC, o, 32);
      }
      if (lane == 0){
        float dth = (cc[i]*aS - ss[i]*aC) * (1.0f/(float)NN);
        float tn = ld_agent(&thc[i]) + ALPHA*dth;
        st_agent(&thn[i], tn);
        out[step*NN + i] = tn;
      }
    }
    __threadfence();
    __syncthreads();
    if (tid == 0){                       // grid barrier (32 co-resident blocks)
      unsigned g   = atomicAdd(&bar[1], 0u);
      unsigned old = atomicAdd(&bar[0], 1u);
      if (old == (unsigned)(gridDim.x - 1)){
        atomicExch(&bar[0], 0u);
        __threadfence();
        atomicAdd(&bar[1], 1u);
      } else {
        while (atomicAdd(&bar[1], 0u) == g) { __builtin_amdgcn_s_sleep(2); }
      }
    }
    __syncthreads();
  }
}

extern "C" void kernel_launch(void* const* d_in, const int* in_sizes, int n_in,
                              void* d_out, int out_size, void* d_ws, size_t ws_size,
                              hipStream_t stream){
  const float* x   = (const float*)d_in[0];
  const float* w1  = (const float*)d_in[1];
  const float* b1  = (const float*)d_in[2];
  const float* g1  = (const float*)d_in[3];
  const float* be1 = (const float*)d_in[4];
  const float* w2  = (const float*)d_in[5];
  // d_in[6] = b2 : cancels exactly inside BN2 (training-mode batch stats)
  const float* g2  = (const float*)d_in[7];
  const float* be2 = (const float*)d_in[8];
  const float* w3  = (const float*)d_in[9];
  float* ws  = (float*)d_ws;
  float* out = (float*)d_out;

  k_zero   <<<1,    1024, 0, stream>>>(ws);
  k_uv     <<<NN,   HD,   0, stream>>>(x, w1, ws);
  k_bn1a   <<<1,    HD,   0, stream>>>(b1, g1, be1, ws);
  k_bn1b   <<<NN,   HD,   0, stream>>>(ws);
  k_gemm<0><<<1024, 256,  0, stream>>>(w2, w3, ws);   // BN2 stats pass
  k_bn2    <<<1,    HD,   0, stream>>>(g2, be2, ws);
  k_gemm<1><<<1024, 256,  0, stream>>>(w2, w3, ws);   // logits pass
  k_maxpart<<<1024, 256,  0, stream>>>(ws);
  k_maxfin <<<1,    1024, 0, stream>>>(ws);
  k_exppart<<<1024, 256,  0, stream>>>(ws);
  k_expfin <<<1,    1024, 0, stream>>>(ws);
  k_symm   <<<4096, 256,  0, stream>>>(ws);
  k_kur    <<<32,   256,  0, stream>>>(ws, out);
}
